// CILRSModel_5669356836352
// MI455X (gfx1250) — matrix operand order repaired and checked by reading, not compiled, not run on hardware
//
#include <hip/hip_runtime.h>

typedef __attribute__((ext_vector_type(16))) _Float16 v16h;
typedef __attribute__((ext_vector_type(8)))  _Float16 v8h;
typedef __attribute__((ext_vector_type(8)))  float    v8f;

#define WMMA(a,b,c) __builtin_amdgcn_wmma_f32_16x16x32_f16(false,(a),false,(b),(short)0,(c),false,false)

__device__ inline v8f vzero8() {
  v8f z;
#pragma unroll
  for (int i = 0; i < 8; ++i) z[i] = 0.f;
  return z;
}

// ---------------------------------------------------------------------------
// k_prep: cast embedding->f16, compute h1 = relu(speed*sW1+sb1) (rank-1),
// and pre-swizzle every weight matrix into WMMA B-fragment order:
//   frag(head, ntile, kchunk)[lane] = 16 contiguous f16:
//     n = ntile*16 + lane%16 ; k = kchunk*32 + (lane/16)*16 + h  (h = 0..15)
// so staging a B-fragment chunk is a contiguous, coalesced burst from L2.
// ---------------------------------------------------------------------------
__global__ __launch_bounds__(256) void k_prep(
    const float* __restrict__ emb32, const float* __restrict__ speed,
    const float* __restrict__ sW1, const float* __restrict__ sb1,
    const float* __restrict__ sW2,
    const float* __restrict__ oW1, const float* __restrict__ ob1,
    const float* __restrict__ oW2, const float* __restrict__ ob2,
    const float* __restrict__ oW3, const float* __restrict__ ob3,
    const float* __restrict__ bW1, const float* __restrict__ bb1,
    const float* __restrict__ bW2, const float* __restrict__ bb2,
    const float* __restrict__ bW3, const float* __restrict__ bb3,
    _Float16* __restrict__ embh, _Float16* __restrict__ h1,
    _Float16* __restrict__ W1f, _Float16* __restrict__ W2f,
    _Float16* __restrict__ W3f, _Float16* __restrict__ sW2f,
    float* __restrict__ b1c, float* __restrict__ b2c, float* __restrict__ b3c)
{
  const long long tid = (long long)blockIdx.x * blockDim.x + threadIdx.x;
  const long long nth = (long long)gridDim.x * blockDim.x;

  for (long long i = tid; i < 65536LL * 512; i += nth) {
    long long b = i >> 9; int e = (int)(i & 511);
    embh[b * 1024 + e] = (_Float16)emb32[i];
  }
  for (long long i = tid; i < 65536LL * 256; i += nth) {
    long long b = i >> 8; int h = (int)(i & 255);
    float x = speed[b] * sW1[h] + sb1[h];
    h1[i] = (_Float16)(x > 0.f ? x : 0.f);
  }
  // W1f: [7][nt16][kc32][lane32][16]
  for (long long i = tid; i < 7LL * 16 * 32 * 32 * 16; i += nth) {
    int hh = (int)(i & 15), lane = (int)((i >> 4) & 31);
    int kc = (int)((i >> 9) & 31), nt = (int)((i >> 14) & 15), head = (int)(i >> 18);
    int n = nt * 16 + (lane & 15);
    int k = kc * 32 + (lane >> 4) * 16 + hh;
    float w = (head < 6) ? bW1[((long long)head * 1024 + k) * 256 + n]
                         : oW1[(long long)k * 256 + n];
    W1f[i] = (_Float16)w;
  }
  // W2f: [7][nt16][kc8][lane32][16]
  for (long long i = tid; i < 7LL * 16 * 8 * 32 * 16; i += nth) {
    int hh = (int)(i & 15), lane = (int)((i >> 4) & 31);
    int kc = (int)((i >> 9) & 7), nt = (int)((i >> 12) & 15), head = (int)(i >> 16);
    int n = nt * 16 + (lane & 15);
    int k = kc * 32 + (lane >> 4) * 16 + hh;
    float w = (head < 6) ? bW2[((long long)head * 256 + k) * 256 + n]
                         : oW2[(long long)k * 256 + n];
    W2f[i] = (_Float16)w;
  }
  // W3f: [7][kc8][lane32][16]  (N padded 3/1 -> 16, zero filled)
  for (long long i = tid; i < 7LL * 8 * 32 * 16; i += nth) {
    int hh = (int)(i & 15), lane = (int)((i >> 4) & 31);
    int kc = (int)((i >> 9) & 7), head = (int)(i >> 12);
    int n = lane & 15;
    int k = kc * 32 + (lane >> 4) * 16 + hh;
    float w = 0.f;
    if (head < 6) { if (n < 3) w = bW3[((long long)head * 256 + k) * 3 + n]; }
    else          { if (n == 0) w = oW3[k]; }
    W3f[i] = (_Float16)w;
  }
  // sW2f: [nt32][kc8][lane32][16]
  for (long long i = tid; i < 32LL * 8 * 32 * 16; i += nth) {
    int hh = (int)(i & 15), lane = (int)((i >> 4) & 31);
    int kc = (int)((i >> 9) & 7), nt = (int)((i >> 12) & 31);
    int n = nt * 16 + (lane & 15);
    int k = kc * 32 + (lane >> 4) * 16 + hh;
    sW2f[i] = (_Float16)sW2[(long long)k * 512 + n];
  }
  for (long long i = tid; i < 7 * 256; i += nth) {
    int head = (int)(i >> 8), n = (int)(i & 255);
    b1c[i] = (head < 6) ? bb1[head * 256 + n] : ob1[n];
    b2c[i] = (head < 6) ? bb2[head * 256 + n] : ob2[n];
  }
  for (long long i = tid; i < 7 * 16; i += nth) {
    int head = (int)(i >> 4), n = (int)(i & 15);
    float v = 0.f;
    if (head < 6) { if (n < 3) v = bb3[head * 3 + n]; }
    else          { if (n == 0) v = ob3[0]; }
    b3c[i] = v;
  }
}

// ---------------------------------------------------------------------------
// k_sp: sp = h1[B,256] @ sW2[256,512] + sb2  ->  emb_f16[:, 512:1024]
// A and B fragments staged through LDS (32 KB total -> high occupancy).
// ---------------------------------------------------------------------------
__global__ __launch_bounds__(256) void k_sp(
    const _Float16* __restrict__ h1,
    const _Float16* __restrict__ sW2f,
    const float* __restrict__ sb2,
    _Float16* __restrict__ embh)
{
  __shared__ __align__(32) v8h lds_a[2][8][32][2];   // 16 KB [ks][wave][lane][half]
  __shared__ __align__(32) v8h lds_b[16][64];        // 16 KB [ks*8+nt][chunk16B]
  const int t = threadIdx.x;
  const int wave = t >> 5, lane = t & 31;
  const int mbase = blockIdx.x * 128;
  const int nblock = blockIdx.y;   // 0..3

  v8f acc[8];
#pragma unroll
  for (int i = 0; i < 8; ++i) acc[i] = vzero8();

  for (int ko = 0; ko < 4; ++ko) {        // K chunks of 64
    __syncthreads();
#pragma unroll
    for (int j = 0; j < 2; ++j) {         // A: 64 rows x 64 K
      int c = t + j * 256;
      int r = c >> 3, c8 = c & 7;
      lds_a[c8 >> 2][r >> 4][(r & 15) + (((c8 >> 1) & 1) << 4)][c8 & 1] =
          *(const v8h*)(h1 + (size_t)(mbase + r) * 256 + ko * 64 + c8 * 8);
    }
#pragma unroll
    for (int j = 0; j < 4; ++j) {         // B: 16 frags x 1 KB
      int c = t + j * 256;
      int f = c >> 6, c64 = c & 63;
      int nt = f & 7, ks = f >> 3;
      lds_b[f][c64] = *(const v8h*)(sW2f +
          (size_t)(((nblock * 8 + nt) * 8 + (ko * 2 + ks)) * 32) * 16 + c64 * 8);
    }
    __syncthreads();
#pragma unroll
    for (int ks = 0; ks < 2; ++ks) {
      v16h a = *(const v16h*)&lds_a[ks][wave][lane][0];
#pragma unroll
      for (int nt = 0; nt < 8; ++nt) {
        v16h b = *(const v16h*)&lds_b[ks * 8 + nt][lane * 2];
        acc[nt] = WMMA(a, b, acc[nt]);
      }
    }
  }
#pragma unroll
  for (int nt = 0; nt < 8; ++nt) {
    int n = nblock * 128 + nt * 16 + (lane & 15);
    float bias = sb2[n];
#pragma unroll
    for (int v = 0; v < 8; ++v) {
      int row = mbase + wave * 16 + v + 8 * (lane >> 4);
      embh[(size_t)row * 1024 + 512 + n] = (_Float16)(acc[nt][v] + bias);
    }
  }
}

// ---------------------------------------------------------------------------
// k_heads: fused 3-layer MLP for 7 heads. 128 rows per WG.
// N split into two passes of 8 n-tiles (64 acc VGPRs live); loads left in
// the nt loop — this is the form the scheduler pipelines best (descending
// s_wait_dscnt over the first 8-WMMA block, C=0 fold on the first K-chunk).
// LDS = lds_h (64 KB) + lds_b2 (8 KB, L2 B-stage) + 64 KB union region:
//   L1 phase : [0:16K] A-stage, [16K:32K] B-stage (16 frags/k-chunk)
//   post-L2  : [0:64K] h2 buffer (A-fragment order)
// 136 KB total -> 2 workgroups co-resident per WGP = 4 waves/SIMD, which
// interleave over the remaining short LDS-latency waits.
// ---------------------------------------------------------------------------
__global__ __launch_bounds__(256, 4) void k_heads(
    const _Float16* __restrict__ embh,
    const _Float16* __restrict__ W1f,
    const _Float16* __restrict__ W2f,
    const _Float16* __restrict__ W3f,
    const float* __restrict__ b1c,
    const float* __restrict__ b2c,
    const float* __restrict__ b3c,
    const int* __restrict__ cmd,
    float* __restrict__ out_ctrl,
    float* __restrict__ out_speed)
{
  __shared__ __align__(32) _Float16 lds_h[8][8][32][16];   // 64 KB [kc][wave][lane][h]
  __shared__ __align__(32) unsigned char lds_u[65536];     // 64 KB union
  __shared__ __align__(32) v8h lds_b2[8][64];              // 8 KB dedicated L2 B-stage

  v8h* UA  = (v8h*)lds_u;                 // [((ks*8+w)*32+lane)*2 + half]
  v8h* UB1 = (v8h*)(lds_u + 16384);       // [(ks*8+nt8)*64 + chunk]
  _Float16* UH2 = (_Float16*)lds_u;       // [((kc*8+w)*32+lane)*16 + h]

  const int t = threadIdx.x;
  const int wave = t >> 5, lane = t & 31;
  const int mbase = blockIdx.x * 128;

  for (int head = 0; head < 7; ++head) {
    const _Float16* w1h = W1f + head * (16 * 32 * 32 * 16);
    // ------------- Layer 1: [128x1024] x [1024x256], two N-half passes ------
    for (int half = 0; half < 2; ++half) {
      v8f acc[8];
#pragma unroll
      for (int i = 0; i < 8; ++i) acc[i] = vzero8();

      for (int ko = 0; ko < 16; ++ko) {       // K in chunks of 64
        __syncthreads();
        // stage A: 512 x 16B, coalesced 128B per row
#pragma unroll
        for (int j = 0; j < 2; ++j) {
          int c = t + j * 256;
          int r = c >> 3, c8 = c & 7;
          UA[(((c8 >> 2) * 8 + (r >> 4)) * 32 + ((r & 15) + (((c8 >> 1) & 1) << 4))) * 2 +
             (c8 & 1)] =
              *(const v8h*)(embh + (size_t)(mbase + r) * 1024 + ko * 64 + c8 * 8);
        }
        // stage B: 16 frags (this N-half) x 1 KB contiguous, dst linear == chunk
#pragma unroll
        for (int j = 0; j < 4; ++j) {
          int c = t + j * 256;
          int f = c >> 6, c64 = c & 63;
          int nt = (f & 7) + half * 8, ks = f >> 3;
          UB1[c] = *(const v8h*)(w1h +
              (size_t)((nt * 32 + (ko * 2 + ks)) * 32) * 16 + c64 * 8);
        }
        // prefetch next K-chunk's source lines into cache while we compute
        if (ko + 1 < 16) {
          int ntp = (t >> 5) + half * 8, sub = t & 31;
          __builtin_prefetch(w1h + (size_t)((ntp * 32 + (ko + 1) * 2) * 512) + sub * 32, 0, 0);
          if (half == 0 && t < 128)
            __builtin_prefetch(embh + (size_t)(mbase + t) * 1024 + (ko + 1) * 64, 0, 0);
        }
        __syncthreads();
#pragma unroll
        for (int ks = 0; ks < 2; ++ks) {
          v16h a = *(const v16h*)&UA[((ks * 8 + wave) * 32 + lane) * 2];
#pragma unroll
          for (int nt8 = 0; nt8 < 8; ++nt8) {
            v16h b = *(const v16h*)&UB1[(ks * 8 + nt8) * 64 + lane * 2];
            acc[nt8] = WMMA(a, b, acc[nt8]);
          }
        }
      }
      // epilogue 1 (this half): bias + relu -> lds_h, A-fragment order
#pragma unroll
      for (int nt8 = 0; nt8 < 8; ++nt8) {
        int nt = half * 8 + nt8;
        float bias = b1c[head * 256 + nt * 16 + (lane & 15)];
        int hh = lane & 15;
        int lbase = 8 * (lane >> 4) + ((nt & 1) << 4);
#pragma unroll
        for (int v = 0; v < 8; ++v) {
          float x = acc[nt8][v] + bias;
          lds_h[nt >> 1][wave][lbase + v][hh] = (_Float16)(x > 0.f ? x : 0.f);
        }
      }
    }
    // ------------- Layer 2: [128x256] x [256x256], two N-half passes --------
    const _Float16* w2h = W2f + head * (16 * 8 * 32 * 16);
    for (int half = 0; half < 2; ++half) {
      v8f acc2[8];
#pragma unroll
      for (int i = 0; i < 8; ++i) acc2[i] = vzero8();

      for (int kc = 0; kc < 8; ++kc) {
        __syncthreads();
#pragma unroll
        for (int j = 0; j < 2; ++j) {       // stage 8 frags x 1 KB
          int c = t + j * 256;
          int f = c >> 6, c64 = c & 63;
          int nt = f + half * 8;
          ((v8h*)lds_b2)[c] = *(const v8h*)(w2h + (size_t)((nt * 8 + kc) * 32) * 16 + c64 * 8);
        }
        __syncthreads();
        v16h a = *(const v16h*)&lds_h[kc][wave][lane][0];
#pragma unroll
        for (int nt8 = 0; nt8 < 8; ++nt8) {
          v16h b = *(const v16h*)&lds_b2[nt8][lane * 2];
          acc2[nt8] = WMMA(a, b, acc2[nt8]);
        }
      }
      // epilogue 2 (this half): bias + relu -> UH2 (union now safe: every wave
      // passed the kc=7 staging barrier, so all L1 UA/UB1 reads are complete)
#pragma unroll
      for (int nt8 = 0; nt8 < 8; ++nt8) {
        int nt = half * 8 + nt8;
        float bias = b2c[head * 256 + nt * 16 + (lane & 15)];
        int hh = lane & 15;
        int lbase = 8 * (lane >> 4) + ((nt & 1) << 4);
#pragma unroll
        for (int v = 0; v < 8; ++v) {
          float x = acc2[nt8][v] + bias;
          UH2[(((nt >> 1) * 8 + wave) * 32 + (lbase + v)) * 16 + hh] =
              (_Float16)(x > 0.f ? x : 0.f);
        }
      }
    }
    // ------------- Layer 3: [128x256] x [256x16(pad)] -------------
    v8f acc3 = vzero8();
    const _Float16* w3h = W3f + head * (8 * 32 * 16);
#pragma unroll
    for (int kc = 0; kc < 8; ++kc) {
      v16h a = *(const v16h*)&UH2[((kc * 8 + wave) * 32 + lane) * 16];
      v16h b = *(const v16h*)(w3h + ((kc * 32 + lane) << 4));
      acc3 = WMMA(a, b, acc3);
    }
    int n = lane & 15;
    if (head < 6) {
      if (n < 3) {
#pragma unroll
        for (int v = 0; v < 8; ++v) {
          int rloc = wave * 16 + v + 8 * (lane >> 4);
          if (cmd[mbase + rloc] == head + 1) {
            float x = acc3[v] + b3c[head * 16 + n];
            out_ctrl[(size_t)(mbase + rloc) * 3 + n] = 1.f / (1.f + __expf(-x));
          }
        }
      }
    } else {
      if (n == 0) {
#pragma unroll
        for (int v = 0; v < 8; ++v) {
          int rloc = wave * 16 + v + 8 * (lane >> 4);
          out_speed[mbase + rloc] = acc3[v] + b3c[6 * 16];
        }
      }
    }
  }
}

// ---------------------------------------------------------------------------
extern "C" void kernel_launch(void* const* d_in, const int* in_sizes, int n_in,
                              void* d_out, int out_size, void* d_ws, size_t ws_size,
                              hipStream_t stream) {
  (void)in_sizes; (void)n_in; (void)out_size; (void)ws_size;

  const float* embedding = (const float*)d_in[0];
  const float* speed     = (const float*)d_in[1];
  const int*   command   = (const int*)d_in[2];
  const float* sW1 = (const float*)d_in[3];
  const float* sb1 = (const float*)d_in[4];
  const float* sW2 = (const float*)d_in[5];
  const float* sb2 = (const float*)d_in[6];
  const float* oW1 = (const float*)d_in[7];
  const float* ob1 = (const float*)d_in[8];
  const float* oW2 = (const float*)d_in[9];
  const float* ob2 = (const float*)d_in[10];
  const float* oW3 = (const float*)d_in[11];
  const float* ob3 = (const float*)d_in[12];
  const float* bW1 = (const float*)d_in[13];
  const float* bb1 = (const float*)d_in[14];
  const float* bW2 = (const float*)d_in[15];
  const float* bb2 = (const float*)d_in[16];
  const float* bW3 = (const float*)d_in[17];
  const float* bb3 = (const float*)d_in[18];

  // workspace layout (bytes), all offsets 256-aligned; total ~164.7 MiB
  char* ws = (char*)d_ws;
  _Float16* embh = (_Float16*)(ws + 0);           // [65536][1024] f16
  _Float16* h1   = (_Float16*)(ws + 134217728);   // [65536][256]  f16
  _Float16* W1f  = (_Float16*)(ws + 167772160);   // 3,670,016 B
  _Float16* W2f  = (_Float16*)(ws + 171442176);   //   917,504 B
  _Float16* W3f  = (_Float16*)(ws + 172359680);   //    57,344 B
  _Float16* sW2f = (_Float16*)(ws + 172417024);   //   262,144 B
  float* b1c = (float*)(ws + 172679168);          //     7,168 B
  float* b2c = (float*)(ws + 172686336);          //     7,168 B
  float* b3c = (float*)(ws + 172693504);          //       448 B

  float* out_ctrl  = (float*)d_out;                // [65536][3]
  float* out_speed = out_ctrl + (size_t)65536 * 3; // [65536][1]

  k_prep<<<4096, 256, 0, stream>>>(embedding, speed, sW1, sb1, sW2,
      oW1, ob1, oW2, ob2, oW3, ob3, bW1, bb1, bW2, bb2, bW3, bb3,
      embh, h1, W1f, W2f, W3f, sW2f, b1c, b2c, b3c);

  k_sp<<<dim3(512, 4), 256, 0, stream>>>(h1, sW2f, sb2, embh);

  k_heads<<<512, 256, 0, stream>>>(embh, W1f, W2f, W3f, b1c, b2c, b3c,
      command, out_ctrl, out_speed);
}
